// align_net_w_feat_33947421508513
// MI455X (gfx1250) — matrix operand order repaired and checked by reading, not compiled, not run on hardware
//
#include <hip/hip_runtime.h>
#include <hip/hip_bf16.h>
#include <stdint.h>

// ---------------------------------------------------------------------------
// MI455X (gfx1250) implementation notes
//  - All GEMM-like work (implicit-GEMM 3x3 convs, deformable-conv einsum)
//    runs on v_wmma_f32_16x16x32_bf16 (bf16 operands, fp32 accumulate).
//  - Activations stored HWC bf16 so WMMA A-fragments are contiguous b128 loads.
//  - Weights pre-packed on device into the ISA B-fragment layout.
//  - Deform conv: bilinear sample -> LDS (32px x 576K bf16 = 36KB) -> WMMA.
// ---------------------------------------------------------------------------

#define HH 128
#define WW 128
#define HW (HH * WW)

typedef __attribute__((ext_vector_type(16))) __bf16 bf16x16;
typedef __attribute__((ext_vector_type(8)))  float  f32x8;

struct B32 { uint4 lo, hi; };

__device__ __forceinline__ float bf2f(unsigned short u) {
  unsigned int v = ((unsigned int)u) << 16;
  return __builtin_bit_cast(float, v);
}
__device__ __forceinline__ unsigned short f2bf(float f) {
  unsigned int x = __builtin_bit_cast(unsigned int, f);
  x += 0x7FFFu + ((x >> 16) & 1u);           // round-to-nearest-even
  return (unsigned short)(x >> 16);
}
__device__ __forceinline__ bf16x16 mkfrag(uint4 a, uint4 b) {
  B32 u; u.lo = a; u.hi = b;
  return __builtin_bit_cast(bf16x16, u);
}

// ---------------------------------------------------------------------------
// Weight packing into WMMA B-fragment layout.
// Packed layout: [kblock*NT + nt][lane 0..31][elem 0..15] bf16, where
//   lanes 0-15 : N = nt*16 + lane,      K(elem e) = e<8 ? e : 8+e       (K 0..7,16..23)
//   lanes 16-31: N = nt*16 + lane - 16, K(elem e) = e<8 ? 8+e : 16+e    (K 8..15,24..31)
// mode 0 (conv):  kblock = tap*KC + kc,  K-within-tap = input channel
// mode 1 (dconv): kblock = kc (K=576),   K = ci*9 + tap  (einsum ogck order)
// ---------------------------------------------------------------------------
__global__ void pack_w(const float* __restrict__ w, unsigned short* __restrict__ dst,
                       int O_real, int I_real, int KC, int NT, int mode, int total)
{
  int idx = blockIdx.x * 256 + threadIdx.x;
  if (idx >= total) return;
  int e    = idx & 15;
  int lane = (idx >> 4) & 31;
  int blk  = idx >> 9;
  int nt   = blk % NT;
  int kblk = blk / NT;
  int half = lane >> 4;
  int o      = nt * 16 + (lane & 15);
  int klocal = (e < 8) ? (half * 8 + e) : (16 + half * 8 + (e - 8));
  int ci, tap;
  if (mode == 0) { tap = kblk / KC; ci = (kblk % KC) * 32 + klocal; }
  else           { int K = kblk * 32 + klocal; ci = K / 9; tap = K - ci * 9; }
  float v = 0.f;
  if (o < O_real && ci < I_real) v = w[((size_t)o * I_real + ci) * 9 + tap];
  dst[idx] = f2bf(v);
}

// x: [10][3][HW] f32 NCHW  ->  xpad: [10][HW][32] bf16 (channels 3..31 zero)
__global__ void pack_input(const float* __restrict__ x, unsigned short* __restrict__ xp)
{
  int idx = blockIdx.x * 256 + threadIdx.x;
  if (idx >= 10 * HW) return;
  int img = idx >> 14;
  int p   = idx & (HW - 1);
  const float* xb = x + (size_t)img * 3 * HW;
  unsigned short* d = xp + (size_t)idx * 32;
  d[0] = f2bf(xb[p]);
  d[1] = f2bf(xb[HW + p]);
  d[2] = f2bf(xb[2 * HW + p]);
#pragma unroll
  for (int c = 3; c < 32; ++c) d[c] = 0;
}

// cat[j][p][0..63] = ref(center) features, [64..127] = supp features
__global__ void concat_cat(const unsigned short* __restrict__ f10,
                           unsigned short* __restrict__ cat)
{
  int idx = blockIdx.x * 256 + threadIdx.x;      // one uint4 (8 bf16) per thread
  if (idx >= 8 * HW * 16) return;
  int c8 = idx & 15;
  int p  = (idx >> 4) & (HW - 1);
  int j  = idx >> 18;
  int b = j >> 2, s = j & 3;
  const unsigned short* srcp;
  if (c8 < 8) srcp = f10 + ((size_t)(b * 5 + 2) * HW + p) * 64 + c8 * 8;
  else        srcp = f10 + ((size_t)(b * 5 + (s < 2 ? s : s + 1)) * HW + p) * 64 + (c8 - 8) * 8;
  *(uint4*)(cat + ((size_t)j * HW + p) * 128 + c8 * 8) = *(const uint4*)srcp;
}

// ---------------------------------------------------------------------------
// Implicit-GEMM 3x3 SAME conv via WMMA.
//   in:  [img][HW][CINP] bf16 (HWC) ; wpack: packed B-fragments
//   block = 256 threads = 8 waves = one image row (M = 128 pixels)
//   wave w: pixels [w*16, w*16+16), all NT 16-wide output-channel tiles
// A-fragment per ISA layout: lane<16 -> M=lane, K {0..7}+{16..23};
//                            lane>=16 -> M=lane-16, K {8..15}+{24..31}
// ---------------------------------------------------------------------------
template<int CINP, int NT, bool RELU, bool RES, bool OUTF32>
__global__ __launch_bounds__(256)
void conv3x3_wmma(const unsigned short* __restrict__ in,
                  const unsigned short* __restrict__ wpack,
                  const float* __restrict__ bias,
                  const unsigned short* __restrict__ res,   // HWC stride NT*16
                  unsigned short* __restrict__ out_bf,
                  float* __restrict__ out_f,
                  int cout_real)
{
  constexpr int KC = CINP / 32;
  constexpr int CO = NT * 16;
  const int h    = blockIdx.x;
  const int img  = blockIdx.y;
  const int wave = threadIdx.x >> 5;
  const int lane = threadIdx.x & 31;
  const int mi   = lane & 15;
  const int half = lane >> 4;
  const int wbase = wave * 16;
  const unsigned short* ibase = in + (size_t)img * HW * CINP;

  f32x8 acc[NT];
  f32x8 zz = {0.f, 0.f, 0.f, 0.f, 0.f, 0.f, 0.f, 0.f};
#pragma unroll
  for (int t = 0; t < NT; ++t) acc[t] = zz;

#pragma unroll
  for (int tap = 0; tap < 9; ++tap) {
    const int dy = tap / 3 - 1, dx = tap % 3 - 1;
    const int row = h + dy;
    const int col = wbase + mi + dx;
    const bool ok = (row >= 0) && (row < HH) && (col >= 0) && (col < WW);
#pragma unroll
    for (int kc = 0; kc < KC; ++kc) {
      uint4 a0 = make_uint4(0, 0, 0, 0), a1 = make_uint4(0, 0, 0, 0);
      if (ok) {
        const unsigned short* ap = ibase + ((size_t)(row * WW + col)) * CINP + kc * 32 + half * 8;
        a0 = *(const uint4*)ap;
        a1 = *(const uint4*)(ap + 16);
      }
      bf16x16 A = mkfrag(a0, a1);
      const int kb = (tap * KC + kc) * NT;
#pragma unroll
      for (int nt = 0; nt < NT; ++nt) {
        const unsigned short* q = wpack + ((size_t)(kb + nt) * 32 + lane) * 16;
        bf16x16 Bm = mkfrag(*(const uint4*)q, *(const uint4*)(q + 8));
        acc[nt] = __builtin_amdgcn_wmma_f32_16x16x32_bf16(
            false, A, false, Bm, (short)0, acc[nt], false, false);
      }
    }
  }

  // Epilogue. C/D layout: VGPR r, lane L -> M = r + 8*(L>=16), N = L&15
#pragma unroll
  for (int nt = 0; nt < NT; ++nt) {
    const int ch = nt * 16 + mi;
    float bv = 0.f;
    if (bias != nullptr && ch < cout_real) bv = bias[ch];
#pragma unroll
    for (int r = 0; r < 8; ++r) {
      const int m = r + half * 8;
      const size_t pix = (size_t)img * HW + h * WW + wbase + m;
      float v = acc[nt][r] + bv;
      if (RELU) v = v > 0.f ? v : 0.f;
      if (RES)  v += bf2f(res[pix * CO + ch]);
      if (OUTF32) out_f[pix * CO + ch] = v;
      else        out_bf[pix * CO + ch] = f2bf(v);
    }
  }
}

// ---------------------------------------------------------------------------
// Deformable conv: bilinear sample 32 pixels x (G=8,Cpg=8,K2=9) = K 576 into
// LDS (bf16), then WMMA GEMM vs packed dconv weights (K-order ci*9+tap).
// grid = (H*4, 8 images); block = 256 threads; each block = 32-pixel segment.
// ---------------------------------------------------------------------------
__global__ __launch_bounds__(256)
void deform_wmma(const unsigned short* __restrict__ src, int src_is_10,
                 const float* __restrict__ offs,          // [8][HW][144] f32
                 const unsigned short* __restrict__ wpack, // [18*4][32][16]
                 unsigned short* __restrict__ out)        // [8][HW][64] bf16
{
  __shared__ unsigned short smem[32 * 576];               // 36 KB
  const int h   = blockIdx.x >> 2;
  const int seg = blockIdx.x & 3;
  const int j   = blockIdx.y;
  const int b = j >> 2, s = j & 3;
  const int simg = src_is_10 ? (b * 5 + (s < 2 ? s : s + 1)) : j;
  const unsigned short* sb = src + (size_t)simg * HW * 64;
  const float* ob = offs + (size_t)j * HW * 144;

  const int tid = threadIdx.x;
  const int pl  = tid & 31;          // local pixel in segment
  const int cc  = tid >> 5;          // channel-in-group 0..7
  const int p   = seg * 32 + pl;     // column
  const int pix = h * WW + p;

  for (int g = 0; g < 8; ++g) {
    const int ch = g * 8 + cc;
#pragma unroll
    for (int k = 0; k < 9; ++k) {
      const float2 dd = *(const float2*)(ob + (size_t)pix * 144 + (g * 9 + k) * 2);
      const float py = dd.x + (float)h + (float)(k / 3 - 1);
      const float px = dd.y + (float)p + (float)(k % 3 - 1);
      const float y0f = floorf(py), x0f = floorf(px);
      const float wy = py - y0f, wx = px - x0f;
      const int y0 = (int)y0f, x0 = (int)x0f;
      float v = 0.f;
#pragma unroll
      for (int cy = 0; cy < 2; ++cy) {
#pragma unroll
        for (int cx = 0; cx < 2; ++cx) {
          const int yc = y0 + cy, xc = x0 + cx;
          const float m = (yc >= 0 && yc < HH && xc >= 0 && xc < WW) ? 1.f : 0.f;
          const float wgt = (cy ? wy : 1.f - wy) * (cx ? wx : 1.f - wx) * m;
          const int yi = yc < 0 ? 0 : (yc > HH - 1 ? HH - 1 : yc);
          const int xi = xc < 0 ? 0 : (xc > WW - 1 ? WW - 1 : xc);
          v += wgt * bf2f(sb[((size_t)(yi * WW + xi)) * 64 + ch]);
        }
      }
      smem[pl * 576 + ch * 9 + k] = f2bf(v);
    }
  }
  __syncthreads();

  // GEMM: 8 waves; wave w -> M-subtile (w&1), N-tile (w>>1). K = 18 chunks.
  const int wave = tid >> 5, lane = tid & 31;
  const int mi = lane & 15, half = lane >> 4;
  const int msub = wave & 1, nt = wave >> 1;
  const int pm = msub * 16 + mi;
  f32x8 acc = {0.f, 0.f, 0.f, 0.f, 0.f, 0.f, 0.f, 0.f};
#pragma unroll
  for (int kc = 0; kc < 18; ++kc) {
    const unsigned short* ap = smem + pm * 576 + kc * 32 + half * 8;
    bf16x16 A = mkfrag(*(const uint4*)ap, *(const uint4*)(ap + 16));
    const unsigned short* q = wpack + ((size_t)(kc * 4 + nt) * 32 + lane) * 16;
    bf16x16 Bm = mkfrag(*(const uint4*)q, *(const uint4*)(q + 8));
    acc = __builtin_amdgcn_wmma_f32_16x16x32_bf16(
        false, A, false, Bm, (short)0, acc, false, false);
  }
  const int ch = nt * 16 + mi;
#pragma unroll
  for (int r = 0; r < 8; ++r) {
    const int m = r + half * 8;
    out[((size_t)j * HW + h * WW + seg * 32 + msub * 16 + m) * 64 + ch] = f2bf(acc[r]);
  }
}

// lrs: [2][5][3][HW] f32 = recon (bf16 HWC stride 16) for supp frames, x for center
__global__ void export_lrs(const unsigned short* __restrict__ rec,
                           const float* __restrict__ x, float* __restrict__ out)
{
  int idx = blockIdx.x * 256 + threadIdx.x;
  if (idx >= 2 * 5 * 3 * HW) return;
  int p = idx & (HW - 1);
  int t = idx >> 14;
  int c = t % 3;
  int bf5 = t / 3;
  int f = bf5 % 5, b = bf5 / 5;
  float v;
  if (f == 2) v = x[idx];                                   // identical layout
  else {
    int s = f < 2 ? f : f - 1;
    v = bf2f(rec[((size_t)(b * 4 + s) * HW + p) * 16 + c]);
  }
  out[idx] = v;
}

// feats: [2][5][64][HW] f32 = fea3 for supp frames, resblock-out for center
__global__ void export_feats(const unsigned short* __restrict__ fea,
                             const unsigned short* __restrict__ f10,
                             float* __restrict__ out)
{
  int idx = blockIdx.x * 256 + threadIdx.x;
  if (idx >= 2 * 5 * 64 * HW) return;
  int p = idx & (HW - 1);
  int t = idx >> 14;
  int c = t & 63;
  int bf5 = t >> 6;
  int f = bf5 % 5, b = bf5 / 5;
  float v;
  if (f == 2) v = bf2f(f10[((size_t)(b * 5 + 2) * HW + p) * 64 + c]);
  else {
    int s = f < 2 ? f : f - 1;
    v = bf2f(fea[((size_t)(b * 4 + s) * HW + p) * 64 + c]);
  }
  out[idx] = v;
}

// ---------------------------------------------------------------------------
extern "C" void kernel_launch(void* const* d_in, const int* in_sizes, int n_in,
                              void* d_out, int out_size, void* d_ws, size_t ws_size,
                              hipStream_t stream)
{
  (void)in_sizes; (void)n_in; (void)out_size; (void)ws_size;
  const float* x       = (const float*)d_in[0];
  const float* w_first = (const float*)d_in[1];
  const float* b_first = (const float*)d_in[2];
  const float* rw1     = (const float*)d_in[3];
  const float* rb1     = (const float*)d_in[4];
  const float* rw2     = (const float*)d_in[5];
  const float* rb2     = (const float*)d_in[6];
  const float* crw     = (const float*)d_in[7];
  const float* crb     = (const float*)d_in[8];
  const float* offw    = (const float*)d_in[9];
  const float* offb    = (const float*)d_in[10];
  const float* dcw     = (const float*)d_in[11];
  const float* recw    = (const float*)d_in[12];
  const float* recb    = (const float*)d_in[13];
  float* out = (float*)d_out;

  char* wsp = (char*)d_ws;
  auto alloc = [&](size_t bytes) -> void* {
    void* pp = (void*)wsp;
    wsp += (bytes + 255) & ~(size_t)255;
    return pp;
  };
  unsigned short* xpad = (unsigned short*)alloc((size_t)10 * HW * 32 * 2);
  unsigned short* fa   = (unsigned short*)alloc((size_t)10 * HW * 64 * 2);
  unsigned short* fb   = (unsigned short*)alloc((size_t)10 * HW * 64 * 2);
  unsigned short* ftmp = (unsigned short*)alloc((size_t)10 * HW * 64 * 2);
  unsigned short* cat  = (unsigned short*)alloc((size_t)8 * HW * 128 * 2);
  unsigned short* g0   = (unsigned short*)alloc((size_t)8 * HW * 64 * 2);
  unsigned short* g1   = (unsigned short*)alloc((size_t)8 * HW * 64 * 2);
  float*          offo = (float*)alloc((size_t)8 * HW * 144 * 4);
  unsigned short* rec  = (unsigned short*)alloc((size_t)8 * HW * 16 * 2);

  unsigned short* pw_first = (unsigned short*)alloc((size_t)18432 * 2);
  unsigned short* pw_r1    = (unsigned short*)alloc((size_t)5 * 36864 * 2);
  unsigned short* pw_r2    = (unsigned short*)alloc((size_t)5 * 36864 * 2);
  unsigned short* pw_cr    = (unsigned short*)alloc((size_t)73728 * 2);
  unsigned short* pw_off   = (unsigned short*)alloc((size_t)4 * 82944 * 2);
  unsigned short* pw_dc    = (unsigned short*)alloc((size_t)4 * 36864 * 2);
  unsigned short* pw_rec   = (unsigned short*)alloc((size_t)9216 * 2);

  auto cdiv = [](int a, int b) { return (a + b - 1) / b; };
  dim3 blk(256);
  dim3 g10(HH, 10), g8(HH, 8);
  dim3 gdef(HH * 4, 8);

  // ---- weight packing (tiny, ~1M elements total) ----
  pack_w<<<cdiv(18432, 256), blk, 0, stream>>>(w_first, pw_first, 64, 3, 1, 4, 0, 18432);
  for (int i = 0; i < 5; ++i) {
    pack_w<<<cdiv(36864, 256), blk, 0, stream>>>(rw1 + (size_t)i * 36864, pw_r1 + (size_t)i * 36864, 64, 64, 2, 4, 0, 36864);
    pack_w<<<cdiv(36864, 256), blk, 0, stream>>>(rw2 + (size_t)i * 36864, pw_r2 + (size_t)i * 36864, 64, 64, 2, 4, 0, 36864);
  }
  pack_w<<<cdiv(73728, 256), blk, 0, stream>>>(crw, pw_cr, 64, 128, 4, 4, 0, 73728);
  for (int i = 0; i < 4; ++i) {
    pack_w<<<cdiv(82944, 256), blk, 0, stream>>>(offw + (size_t)i * 82944, pw_off + (size_t)i * 82944, 144, 64, 2, 9, 0, 82944);
    pack_w<<<cdiv(36864, 256), blk, 0, stream>>>(dcw + (size_t)i * 36864, pw_dc + (size_t)i * 36864, 64, 64, 18, 4, 1, 36864);
  }
  pack_w<<<cdiv(9216, 256), blk, 0, stream>>>(recw, pw_rec, 3, 64, 2, 1, 0, 9216);

  pack_input<<<cdiv(10 * HW, 256), blk, 0, stream>>>(x, xpad);

  // ---- feature extraction: conv_first + 5 residual blocks ----
  conv3x3_wmma<32, 4, true, false, false><<<g10, blk, 0, stream>>>(
      xpad, pw_first, b_first, nullptr, fa, nullptr, 64);
  for (int i = 0; i < 5; ++i) {
    conv3x3_wmma<64, 4, true, false, false><<<g10, blk, 0, stream>>>(
        fa, pw_r1 + (size_t)i * 36864, rb1 + (size_t)i * 64, nullptr, ftmp, nullptr, 64);
    conv3x3_wmma<64, 4, false, true, false><<<g10, blk, 0, stream>>>(
        ftmp, pw_r2 + (size_t)i * 36864, rb2 + (size_t)i * 64, fa, fb, nullptr, 64);
    unsigned short* t = fa; fa = fb; fb = t;
  }

  // ---- concat(ref, supp) + cr conv ----
  concat_cat<<<cdiv(8 * HW * 16, 256), blk, 0, stream>>>(fa, cat);
  conv3x3_wmma<128, 4, false, false, false><<<g8, blk, 0, stream>>>(
      cat, pw_cr, crb, nullptr, g0, nullptr, 64);

  // ---- deformable alignment cascade ----
  // stage 0: fea1 = deform(fea0, off0)
  conv3x3_wmma<64, 9, false, false, true><<<g8, blk, 0, stream>>>(
      g0, pw_off, offb, nullptr, nullptr, offo, 144);
  deform_wmma<<<gdef, blk, 0, stream>>>(g0, 0, offo, pw_dc, g1);
  // stage 1: fea2 = deform(fea1, off1)
  conv3x3_wmma<64, 9, false, false, true><<<g8, blk, 0, stream>>>(
      g1, pw_off + (size_t)82944, offb + 144, nullptr, nullptr, offo, 144);
  deform_wmma<<<gdef, blk, 0, stream>>>(g1, 0, offo, pw_dc + (size_t)36864, g0);
  // stage 2: fea3 = deform(supp, off2(fea2))  -- supp lives in the 10-img buffer
  conv3x3_wmma<64, 9, false, false, true><<<g8, blk, 0, stream>>>(
      g0, pw_off + 2 * (size_t)82944, offb + 288, nullptr, nullptr, offo, 144);
  deform_wmma<<<gdef, blk, 0, stream>>>(fa, 1, offo, pw_dc + 2 * (size_t)36864, g1);
  // stage 3: aligned = deform(fea3, off3(fea3))
  conv3x3_wmma<64, 9, false, false, true><<<g8, blk, 0, stream>>>(
      g1, pw_off + 3 * (size_t)82944, offb + 432, nullptr, nullptr, offo, 144);
  deform_wmma<<<gdef, blk, 0, stream>>>(g1, 0, offo, pw_dc + 3 * (size_t)36864, g0);

  // ---- recon conv (64 -> 3, padded to one 16-wide N tile) ----
  conv3x3_wmma<64, 1, false, false, false><<<g8, blk, 0, stream>>>(
      g0, pw_rec, recb, nullptr, rec, nullptr, 3);

  // ---- assemble fp32 NCHW outputs ----
  export_lrs<<<cdiv(2 * 5 * 3 * HW, 256), blk, 0, stream>>>(rec, x, out);
  export_feats<<<cdiv(2 * 5 * 64 * HW, 256), blk, 0, stream>>>(
      g1, fa, out + (size_t)2 * 5 * 3 * HW);
}
